// GORUCell_61134564491659
// MI455X (gfx1250) — matrix operand order repaired and checked by reading, not compiled, not run on hardware
//
#include <hip/hip_runtime.h>
#include <hip/hip_bf16.h>

// ---------------------------------------------------------------------------
// GORU cell for MI455X (gfx1250, wave32, WMMA).
//   k0: build EUNN rotation tables (replicates reference reshape quirks)
//   k1: f16-WMMA GEMM  gates = [X|hx] @ [gate_U ; gate_W] + gate_bias  -> ws
//   k2: f16-WMMA GEMM  Ux    = X @ U                                   -> d_out
//   k3: fused EUNN (16 butterfly layers) + modReLU + gate blend, in-place on d_out
// ---------------------------------------------------------------------------

#define BM 128
#define BN 128
#define BK 32
#define BKP 40   // padded f16 stride: 80 bytes/row, keeps 16B alignment

typedef __attribute__((ext_vector_type(16))) _Float16 v16h;
typedef __attribute__((ext_vector_type(8)))  _Float16 v8h;
typedef __attribute__((ext_vector_type(8)))  float    v8f;

union Frag16 { v16h v; v8h h[2]; };

// ---------------------------------------------------------------------------
// WMMA GEMM: C[M,N] = concat_K(A0,A1) @ concat_K(W0,W1) (+ bias), fp32 io,
// f16 multiply / f32 accumulate.  K tiles < kSplit come from (A0,W0), the
// rest from (A1,W1).  A row strides: kSplit / (K-kSplit). W row stride: N.
// ---------------------------------------------------------------------------
__global__ __launch_bounds__(256)
void gemm_f16_wmma(const float* __restrict__ A0, const float* __restrict__ W0,
                   const float* __restrict__ A1, const float* __restrict__ W1,
                   const float* __restrict__ bias, float* __restrict__ C,
                   int M, int N, int K, int kSplit)
{
    (void)M;
    __shared__ _Float16 As[BM][BKP];   // [m][k]
    __shared__ _Float16 Bs[BN][BKP];   // [n][k]  (transposed for contiguous frags)

    const int tid  = threadIdx.x;
    const int lane = tid & 31;
    const int wave = tid >> 5;      // 0..7
    const int wm   = wave & 1;      // 2 waves along M (64 rows each)
    const int wn   = wave >> 1;     // 4 waves along N (32 cols each)
    const int bm   = blockIdx.y * BM;
    const int bn   = blockIdx.x * BN;
    const int g    = lane >> 4;     // K-group selector per ISA A/B layouts
    const int l16  = lane & 15;

    v8f acc[4][2] = {};

    const int ldA0 = kSplit;
    const int ldA1 = K - kSplit;
    const int numKT = K / BK;

    // cooperative load coords (16 fp32 per thread per tile)
    const int arow  = tid >> 1;          // 0..127
    const int akoff = (tid & 1) * 16;    // 0 / 16
    const int bkoff = tid & 31;          // k within tile
    const int bnoff = (tid >> 5) * 16;   // n segment

    for (int kt = 0; kt < numKT; ++kt) {
        const int kk = kt * BK;
        const float* Ap; const float* Wp; int kkL, ldA;
        if (kk < kSplit) { Ap = A0; Wp = W0; kkL = kk;          ldA = ldA0; }
        else             { Ap = A1; Wp = W1; kkL = kk - kSplit; ldA = ldA1; }

        __syncthreads();
        {   // A tile: 128x32 fp32 -> f16 LDS (row-major)
            const float* src = Ap + (size_t)(bm + arow) * ldA + kkL + akoff;
            const float4* s4 = (const float4*)src;
            #pragma unroll
            for (int v = 0; v < 4; ++v) {
                float4 f = s4[v];
                _Float16* d = &As[arow][akoff + v * 4];
                d[0] = (_Float16)f.x; d[1] = (_Float16)f.y;
                d[2] = (_Float16)f.z; d[3] = (_Float16)f.w;
            }
            if (kt + 1 < numKT) __builtin_prefetch(src + BK, 0, 0);
        }
        {   // B tile: 32x128 fp32 row-major -> transposed f16 LDS [n][k]
            const float* src = Wp + (size_t)(kkL + bkoff) * N + bn + bnoff;
            const float4* s4 = (const float4*)src;
            #pragma unroll
            for (int v = 0; v < 4; ++v) {
                float4 f = s4[v];
                const int nb = bnoff + v * 4;
                Bs[nb + 0][bkoff] = (_Float16)f.x;
                Bs[nb + 1][bkoff] = (_Float16)f.y;
                Bs[nb + 2][bkoff] = (_Float16)f.z;
                Bs[nb + 3][bkoff] = (_Float16)f.w;
            }
            if (kt + 1 < numKT) __builtin_prefetch(src + (size_t)BK * N, 0, 0);
        }
        __syncthreads();

        Frag16 b[2];
        #pragma unroll
        for (int nt = 0; nt < 2; ++nt) {     // B 32x16: lane=N, 16 contiguous K at g*16
            const int col = wn * 32 + nt * 16 + l16;
            b[nt].h[0] = *(const v8h*)&Bs[col][g * 16];
            b[nt].h[1] = *(const v8h*)&Bs[col][g * 16 + 8];
        }
        #pragma unroll
        for (int mt = 0; mt < 4; ++mt) {     // A 16x32: lane=M, 16 contiguous K at g*16
            Frag16 a;
            const int row = wm * 64 + mt * 16 + l16;
            a.h[0] = *(const v8h*)&As[row][g * 16];
            a.h[1] = *(const v8h*)&As[row][g * 16 + 8];
            #pragma unroll
            for (int nt = 0; nt < 2; ++nt) {
                acc[mt][nt] = __builtin_amdgcn_wmma_f32_16x16x32_f16(
                    false, a.v, false, b[nt].v, (short)0, acc[mt][nt], false, false);
            }
        }
    }

    // epilogue: C layout per ISA (VGPR r -> M = r + 8*(lane>=16), N = lane&15)
    const int hi8 = (lane >> 4) * 8;
    #pragma unroll
    for (int mt = 0; mt < 4; ++mt) {
        #pragma unroll
        for (int nt = 0; nt < 2; ++nt) {
            const int col  = bn + wn * 32 + nt * 16 + l16;
            const float bv = bias ? bias[col] : 0.f;
            const int row0 = bm + wm * 64 + mt * 16 + hi8;
            #pragma unroll
            for (int r = 0; r < 8; ++r)
                C[(size_t)(row0 + r) * N + col] = acc[mt][nt][r] + bv;
        }
    }
}

// ---------------------------------------------------------------------------
// EUNN rotation tables, replicating the reference's (N/2, L/2, 2)->(L/2, N)
// C-order reshape (which interleaves the j and l indices).
// tabs layout: diagA | offA | diagB | offB, each [8][2048] fp32.
// ---------------------------------------------------------------------------
__global__ void build_tables(const float* __restrict__ thetaA,
                             const float* __restrict__ thetaB,
                             float* __restrict__ tabs)
{
    const int t = blockIdx.x * blockDim.x + threadIdx.x;
    if (t >= 8 * 2048) return;
    float* diagA = tabs;
    float* offA  = tabs + 16384;
    float* diagB = tabs + 32768;
    float* offB  = tabs + 49152;
    {   // A: flat index t of (8,2048) maps back into (1024,8,2)
        const int j = t >> 4, l = (t >> 1) & 7, p = t & 1;
        const float th = thetaA[j * 8 + l];
        const float s = sinf(th), c = cosf(th);
        diagA[t] = c;
        offA[t]  = p ? s : -s;
    }
    {   // B: columns 0 and 2047 are identity; interior maps into (1023,8,2)
        const int lp = t >> 11, np = t & 2047;
        float d = 1.f, o = 0.f;
        if (np > 0 && np < 2047) {
            const int f = lp * 2046 + (np - 1);
            const int j = f >> 4, l = (f >> 1) & 7, p = f & 1;
            const float th = thetaB[j * 8 + l];
            const float s = sinf(th);
            d = cosf(th);
            o = p ? s : -s;
        }
        diagB[t] = d; offB[t] = o;
    }
}

// ---------------------------------------------------------------------------
// Fused EUNN + modReLU + gate blend. One block per batch row; each thread
// owns 8 contiguous columns, so butterfly layer A is register-local and
// layer B exchanges 2 edge floats via LDS.
// ---------------------------------------------------------------------------
__device__ __forceinline__ void load8(const float* p, float* d) {
    float4 a = ((const float4*)p)[0];
    float4 b = ((const float4*)p)[1];
    d[0]=a.x; d[1]=a.y; d[2]=a.z; d[3]=a.w;
    d[4]=b.x; d[5]=b.y; d[6]=b.z; d[7]=b.w;
}

__global__ __launch_bounds__(256)
void goru_fuse(const float* __restrict__ hx, const float* __restrict__ G,
               const float* __restrict__ tabs, const float* __restrict__ bias,
               float* __restrict__ out)
{
    const int N   = 2048;
    const int row = blockIdx.x;
    const int tid = threadIdx.x;
    const int c0  = tid * 8;

    __shared__ float edgeL[256];
    __shared__ float edgeR[256];

    float x[8], h[8];
    load8(hx + (size_t)row * N + c0, x);
    #pragma unroll
    for (int i = 0; i < 8; ++i) h[i] = x[i];

    const float* diagA = tabs;
    const float* offA  = tabs + 16384;
    const float* diagB = tabs + 32768;
    const float* offB  = tabs + 49152;

    for (int l = 0; l < 8; ++l) {
        const int base = l * 2048 + c0;
        float dA[8], oA[8], dB[8], oB[8];
        load8(diagA + base, dA); load8(offA + base, oA);
        // layer A: rotate pairs (2j, 2j+1) — fully local
        #pragma unroll
        for (int j = 0; j < 4; ++j) {
            const float e = x[2*j], o = x[2*j+1];
            x[2*j]   = e * dA[2*j]   + o * oA[2*j];
            x[2*j+1] = o * dA[2*j+1] + e * oA[2*j+1];
        }
        // layer B: rotate pairs (2j+1, 2j+2) — exchange one float each side
        edgeL[tid] = x[0]; edgeR[tid] = x[7];
        __syncthreads();
        const float lv = (tid > 0)   ? edgeR[tid - 1] : 0.f;  // killed by oB=0 at col 0
        const float rv = (tid < 255) ? edgeL[tid + 1] : 0.f;  // killed by oB=0 at col 2047
        __syncthreads();
        load8(diagB + base, dB); load8(offB + base, oB);
        const float n0 = x[0] * dB[0] + lv * oB[0];
        #pragma unroll
        for (int j = 0; j < 3; ++j) {
            const float a = x[2*j+1], b = x[2*j+2];
            x[2*j+1] = a * dB[2*j+1] + b * oB[2*j+1];
            x[2*j+2] = b * dB[2*j+2] + a * oB[2*j+2];
        }
        const float n7 = x[7] * dB[7] + rv * oB[7];
        x[0] = n0; x[7] = n7;
    }

    // epilogue: newh = Ux + eunn*r; modReLU; out = hx*z + (1-z)*newh
    const float* g = G + (size_t)row * 4096;
    float r8[8], z8[8], u8[8], bs[8];
    load8(g + c0, r8);
    load8(g + 2048 + c0, z8);
    load8(out + (size_t)row * N + c0, u8);   // staged Ux (in-place, same thread)
    load8(bias + c0, bs);

    float o8[8];
    #pragma unroll
    for (int i = 0; i < 8; ++i) {
        const float nh  = u8[i] + x[i] * r8[i];
        float mag = fabsf(nh) + bs[i];
        mag = mag > 0.f ? mag : 0.f;
        const float s = nh > 0.f ? mag : (nh < 0.f ? -mag : 0.f);
        o8[i] = h[i] * z8[i] + (1.f - z8[i]) * s;
    }
    float4* op = (float4*)(out + (size_t)row * N + c0);
    op[0] = make_float4(o8[0], o8[1], o8[2], o8[3]);
    op[1] = make_float4(o8[4], o8[5], o8[6], o8[7]);
}

// ---------------------------------------------------------------------------
extern "C" void kernel_launch(void* const* d_in, const int* in_sizes, int n_in,
                              void* d_out, int out_size, void* d_ws, size_t ws_size,
                              hipStream_t stream)
{
    (void)in_sizes; (void)n_in; (void)out_size; (void)ws_size;
    const float* input_    = (const float*)d_in[0];  // 4096 x 2048
    const float* hx        = (const float*)d_in[1];  // 4096 x 2048
    const float* U         = (const float*)d_in[2];  // 2048 x 2048
    const float* thetaA    = (const float*)d_in[3];  // 1024 x 8
    const float* thetaB    = (const float*)d_in[4];  // 1023 x 8
    const float* bias      = (const float*)d_in[5];  // 2048
    const float* gate_U    = (const float*)d_in[6];  // 2048 x 4096
    const float* gate_W    = (const float*)d_in[7];  // 2048 x 4096
    const float* gate_bias = (const float*)d_in[8];  // 4096
    float* out = (float*)d_out;                      // 4096 x 2048

    float* G    = (float*)d_ws;                      // 4096 x 4096 gates (64 MB)
    float* tabs = G + (size_t)4096 * 4096;           // 4 x 8 x 2048 (256 KB)

    build_tables<<<64, 256, 0, stream>>>(thetaA, thetaB, tabs);

    // gates = [X|hx] @ [gate_U ; gate_W] + gate_bias   (M=4096, N=4096, K=4096)
    gemm_f16_wmma<<<dim3(4096 / BN, 4096 / BM), 256, 0, stream>>>(
        input_, gate_U, hx, gate_W, gate_bias, G, 4096, 4096, 4096, 2048);

    // Ux = X @ U   (M=4096, N=2048, K=2048) -> staged in d_out
    gemm_f16_wmma<<<dim3(2048 / BN, 4096 / BM), 256, 0, stream>>>(
        input_, U, input_, U, nullptr, out, 4096, 2048, 2048, 2048);

    goru_fuse<<<4096, 256, 0, stream>>>(hx, G, tabs, bias, out);
}
